// Attention_42949672960165
// MI455X (gfx1250) — matrix operand order repaired and checked by reading, not compiled, not run on hardware
//
#include <hip/hip_runtime.h>
#include <hip/hip_bf16.h>

typedef __attribute__((ext_vector_type(16))) __bf16 v16bf;
typedef __attribute__((ext_vector_type(8)))  float  v8f;
typedef __attribute__((ext_vector_type(4)))  unsigned int v4u;
typedef __attribute__((ext_vector_type(8)))  int v8i_t;
typedef __attribute__((ext_vector_type(4)))  int v4i_t;
typedef unsigned short ushort_t;

#define USE_TDM 1

// ---------------------------------------------------------------------------
// float -> bf16, round-to-nearest-even
// ---------------------------------------------------------------------------
__device__ __forceinline__ ushort_t f2bf(float f) {
    union { float f; unsigned int u; } x; x.f = f;
    unsigned int u = x.u;
    unsigned int r = u + 0x7FFFu + ((u >> 16) & 1u);
    return (ushort_t)(r >> 16);
}

// Branch-free tanh for x >= 0 (post-ReLU): (1 - e^-2x) / (1 + e^-2x).
__device__ __forceinline__ float tanh_pos(float x) {
    float t = __expf(-2.0f * x);
    return (1.0f - t) / (1.0f + t);
}

// ---------------------------------------------------------------------------
// TDM: async-copy one contiguous 32 KB Wk strip (already bf16 + swizzled)
// from global into LDS.  1-D descriptor: data_size=4B, tile_dim0=8192.
// ---------------------------------------------------------------------------
__device__ __forceinline__ void tdm_load_b_strip(const void* gsrc,
                                                 unsigned int lds_byte_off) {
    unsigned long long ga = (unsigned long long)(size_t)gsrc;
    v4u g0;
    g0[0] = 1u;                                            // count=1 (valid)
    g0[1] = lds_byte_off;                                  // lds_addr
    g0[2] = (unsigned int)(ga & 0xFFFFFFFFu);              // global_addr lo
    g0[3] = (unsigned int)((ga >> 32) & 0x01FFFFFFu)       // global_addr hi
          | (2u << 30);                                    // type=2 ("image")
    v8i_t g1;
    g1[0] = (int)(2u << 16);          // workgroup_mask=0, data_size=2 (4B)
    g1[1] = (int)(0x2000u << 16);     // tensor_dim0[15:0] = 8192 dwords
    g1[2] = (int)(1u << 16);          // tensor_dim0 hi = 0, tensor_dim1 = 1
    g1[3] = (int)(0x2000u << 16);     // tile_dim0 = 8192 dwords
    g1[4] = 1;                        // tile_dim1 = 1, tile_dim2 = 0
    g1[5] = 8192;                     // tensor_dim0_stride lo
    g1[6] = 0;
    g1[7] = 0;
    v4i_t z = {0, 0, 0, 0};
#if __clang_major__ >= 23
    v8i_t z8 = {0, 0, 0, 0, 0, 0, 0, 0};
    __builtin_amdgcn_tensor_load_to_lds(g0, g1, z, z, z8, 0);
#else
    __builtin_amdgcn_tensor_load_to_lds(g0, g1, z, z, 0);
#endif
}

// ---------------------------------------------------------------------------
// Kernel 0: Wk (f32 [v][d]) -> bf16, pre-swizzled WMMA-B tiles
//   wk_swz[ntile][kc][lane][e];  lane = n%16 + (kl>=16 ? 16:0), e = kl%16.
// ---------------------------------------------------------------------------
__global__ void wk_convert_kernel(const float* __restrict__ Wk,
                                  ushort_t* __restrict__ wk_swz) {
    int idx = blockIdx.x * 256 + threadIdx.x;     // 0 .. 1024*1024-1
    int n = idx & 1023;
    int k = idx >> 10;
    int nt = n >> 4, nl = n & 15;
    int kc = k >> 5, kl = k & 31;
    int ln = nl + ((kl & 16) ? 16 : 0);
    int e  = kl & 15;
    wk_swz[(((nt * 32 + kc) * 32) + ln) * 16 + e] = f2bf(Wk[k * 1024 + n]);
}

// ---------------------------------------------------------------------------
// Kernel 1: q path.  eq[b] = sum_d tanh(relu(query[b]@Wq + bq))[d] * We_q[d]
// ---------------------------------------------------------------------------
__global__ void qpath_kernel(const float* __restrict__ query,
                             const float* __restrict__ Wq,
                             const float* __restrict__ bq,
                             const float* __restrict__ we_q,
                             float* __restrict__ eq) {
    __shared__ float qrow[1024];
    __shared__ float red[256];
    int b = blockIdx.x, tid = threadIdx.x;
    for (int i = tid; i < 1024; i += 256) qrow[i] = query[b * 1024 + i];
    __syncthreads();
    float acc[4] = {0.f, 0.f, 0.f, 0.f};
    for (int v = 0; v < 1024; ++v) {
        float qv = qrow[v];
        const float* wr = Wq + (size_t)v * 1024;
        #pragma unroll
        for (int j = 0; j < 4; ++j) acc[j] = fmaf(qv, wr[tid + j * 256], acc[j]);
    }
    float sum = 0.f;
    #pragma unroll
    for (int j = 0; j < 4; ++j) {
        int dcol = tid + j * 256;
        float val = acc[j] + bq[dcol];
        val = val > 0.f ? val : 0.f;
        sum += tanh_pos(val) * we_q[dcol];
    }
    red[tid] = sum; __syncthreads();
    for (int st = 128; st > 0; st >>= 1) {
        if (tid < st) red[tid] += red[tid + st];
        __syncthreads();
    }
    if (tid == 0) eq[b] = red[0];
}

// ---------------------------------------------------------------------------
// Kernel 2: fused GEMM + energy reduction, double-buffered TDM pipeline.
//   LDS: A (64 rows bf16, WMMA-A swizzled)       : 128 KB @ 0
//        B buffers: [nhalf][parity] 4 x 32 KB    : @ 128K
// Waves 0-3: m-tiles 0-3 on even n-tiles;  waves 4-7: same on odd.
// Strip it+1 streams via TDM while strip it is consumed by WMMA.
// ---------------------------------------------------------------------------
__global__ void energy_kernel(const float* __restrict__ states,
                              const ushort_t* __restrict__ wk_swz,
                              const float* __restrict__ bk,
                              const float* __restrict__ we_k,
                              float* __restrict__ ek2) {
    extern __shared__ char smem[];
    ushort_t* Alds = (ushort_t*)smem;                      // 64K ushorts (128 KB)
    ushort_t* Blds = (ushort_t*)(smem + 128 * 1024);       // 4 x 16K ushorts

    const int tid  = threadIdx.x;
    const int wave = tid >> 5;
    const int lane = tid & 31;
    const int rowBase = blockIdx.x * 64;                   // global row in [0,32768)
    const int b  = rowBase >> 11;                          // /2048
    const int s0 = rowBase & 2047;

    const int nhalf = wave >> 2;                   // 0: even n-tiles, 1: odd
    const int mt    = wave & 3;                    // m-tile of this wave
    const int nl    = lane & 15;

#if USE_TDM
    // Kick off strip 0 immediately: B buffers are disjoint from the A region,
    // so the DMA overlaps the whole A staging phase below.
    if (mt == 0) {
        tdm_load_b_strip(wk_swz + (size_t)nhalf * 16384,
                         131072u + (unsigned)(nhalf * 2) * 32768u);
    }
#endif

    // ---- Phase 1: stage 64x1024 f32 rows -> bf16 WMMA-A swizzled LDS ----
    const float4* src = (const float4*)(states + (size_t)rowBase * 1024);
    #pragma unroll 4
    for (int it = 0; it < 64; ++it) {
        int idx4 = tid + it * 256;                 // float4 index, 16384 total
        int m  = idx4 >> 8;                        // row 0..63
        int kq = (idx4 & 255) << 2;                // first k (multiple of 4)
        float4 v = src[idx4];
        int mti = m >> 4, ml = m & 15;
        int kc = kq >> 5, kl = kq & 31;
        int lhi = (kl >> 4) & 1;
        int e  = kl & 15;                          // e in {0,4,8,12}
        int ln = ml + (lhi ? 16 : 0);
        int off = (((mti * 32 + kc) * 32 + ln) << 4) + e;  // ushort index
        unsigned long long pk =
            (unsigned long long)f2bf(v.x)        |
            ((unsigned long long)f2bf(v.y) << 16) |
            ((unsigned long long)f2bf(v.z) << 32) |
            ((unsigned long long)f2bf(v.w) << 48);
        *(unsigned long long*)(&Alds[off]) = pk;
    }
    __syncthreads();

    // ---- Phase 2: WMMA over n-tiles, TDM double-buffered ----
    const v16bf* Av = (const v16bf*)Alds;
    float eacc[8] = {0.f};

    for (int it = 0; it < 32; ++it) {
        int nt = it * 2 + nhalf;                   // this iteration's n-tile
        if (it > 0) __syncthreads();               // prev readers of target buf done
#if USE_TDM
        if (mt == 0) {
            if (it + 1 < 32) {                     // prefetch next strip
                int ntn = (it + 1) * 2 + nhalf;
                tdm_load_b_strip(wk_swz + (size_t)ntn * 16384,
                                 131072u + (unsigned)(nhalf * 2 + ((it + 1) & 1)) * 32768u);
                __builtin_amdgcn_s_wait_tensorcnt(1);  // strip `it` complete
            } else {
                __builtin_amdgcn_s_wait_tensorcnt(0);
            }
        }
#else
        {   // fallback: cooperative copy of strip `it`
            const uint4* bsrc = (const uint4*)(wk_swz + (size_t)nt * 16384);
            uint4* bdst = (uint4*)(Blds + (nhalf * 2 + (it & 1)) * 16384);
            int lt = mt * 32 + lane;
            #pragma unroll
            for (int j = 0; j < 16; ++j) bdst[lt + j * 128] = bsrc[lt + j * 128];
        }
#endif
        __syncthreads();                           // strip `it` visible to all

        const v16bf* Bv = (const v16bf*)(Blds + (nhalf * 2 + (it & 1)) * 16384);
        v8f c = {};
        #pragma unroll 8
        for (int kc = 0; kc < 32; ++kc) {
            v16bf a  = Av[(mt * 32 + kc) * 32 + lane];
            v16bf bb = Bv[kc * 32 + lane];
            c = __builtin_amdgcn_wmma_f32_16x16x32_bf16(
                    false, a, false, bb, (short)0, c, false, false);
        }

        // Epilogue: lane owns column n = nt*16 + nl for rows v (+8 if lane>=16)
        int ng = nt * 16 + nl;
        float bkn = bk[ng];
        float wkn = we_k[ng];
        #pragma unroll
        for (int v = 0; v < 8; ++v) {
            float val = c[v] + bkn;
            val = val > 0.f ? val : 0.f;
            eacc[v] += tanh_pos(val) * wkn;
        }
    }

    // Reduce the 16 lanes sharing each row group (offsets 1..8 stay in-half)
    #pragma unroll
    for (int off = 1; off < 16; off <<= 1) {
        #pragma unroll
        for (int v = 0; v < 8; ++v) eacc[v] += __shfl_xor(eacc[v], off, 32);
    }

    if (nl == 0) {   // lanes 0 and 16 write rows [0..7] / [8..15] of the m-tile
        int rbase = s0 + mt * 16 + ((lane >> 4) ? 8 : 0);
        float* dst = ek2 + ((size_t)(nhalf * 16 + b)) * 2048;
        #pragma unroll
        for (int v = 0; v < 8; ++v) dst[rbase + v] = eacc[v];
    }
}

// ---------------------------------------------------------------------------
// Kernel 3a: softmax over s for each batch; writes attn_weights to d_out.
// ---------------------------------------------------------------------------
__global__ void softmax_kernel(const float* __restrict__ ek2,
                               const float* __restrict__ eq,
                               const float* __restrict__ be,
                               float* __restrict__ out_w) {
    __shared__ float red[256];
    int b = blockIdx.x, tid = threadIdx.x;
    float base = eq[b] + be[0];
    float e[8];
    float mx = -3.0e38f;
    #pragma unroll
    for (int j = 0; j < 8; ++j) {
        int s = tid + j * 256;
        e[j] = ek2[b * 2048 + s] + ek2[(16 + b) * 2048 + s] + base;
        mx = fmaxf(mx, e[j]);
    }
    red[tid] = mx; __syncthreads();
    for (int st = 128; st > 0; st >>= 1) {
        if (tid < st) red[tid] = fmaxf(red[tid], red[tid + st]);
        __syncthreads();
    }
    mx = red[0]; __syncthreads();
    float sum = 0.f;
    #pragma unroll
    for (int j = 0; j < 8; ++j) { e[j] = __expf(e[j] - mx); sum += e[j]; }
    red[tid] = sum; __syncthreads();
    for (int st = 128; st > 0; st >>= 1) {
        if (tid < st) red[tid] += red[tid + st];
        __syncthreads();
    }
    float inv = 1.f / red[0];
    #pragma unroll
    for (int j = 0; j < 8; ++j) out_w[b * 2048 + tid + j * 256] = e[j] * inv;
}

// ---------------------------------------------------------------------------
// Kernel 3b: partial attn = sum over a 128-row chunk of w[s]*states[b,s,:]
// ---------------------------------------------------------------------------
__global__ void attn_partial_kernel(const float* __restrict__ out_w,
                                    const float* __restrict__ states,
                                    float* __restrict__ part) {
    int b  = blockIdx.x >> 4;
    int ch = blockIdx.x & 15;
    int tid = threadIdx.x;
    float a0 = 0.f, a1 = 0.f, a2 = 0.f, a3 = 0.f;
    int srow = b * 2048 + ch * 128;
    for (int i = 0; i < 128; ++i) {
        float w = out_w[srow + i];
        const float* sp = states + (size_t)(srow + i) * 1024;
        __builtin_prefetch(sp + 2 * 1024, 0, 0);   // global_prefetch, 2 rows ahead
        a0 = fmaf(w, sp[tid],       a0);
        a1 = fmaf(w, sp[tid + 256], a1);
        a2 = fmaf(w, sp[tid + 512], a2);
        a3 = fmaf(w, sp[tid + 768], a3);
    }
    float* pp = part + (size_t)blockIdx.x * 1024;
    pp[tid] = a0; pp[tid + 256] = a1; pp[tid + 512] = a2; pp[tid + 768] = a3;
}

// ---------------------------------------------------------------------------
// Kernel 3c: reduce the 16 chunk partials into attn (second output block).
// ---------------------------------------------------------------------------
__global__ void attn_reduce_kernel(const float* __restrict__ part,
                                   float* __restrict__ attn_out) {
    int idx = blockIdx.x * 256 + threadIdx.x;   // 0..16383
    int b = idx >> 10, dcol = idx & 1023;
    float s = 0.f;
    #pragma unroll
    for (int ch = 0; ch < 16; ++ch) s += part[((size_t)(b * 16 + ch)) * 1024 + dcol];
    attn_out[idx] = s;
}

// ---------------------------------------------------------------------------
// Launch
// ---------------------------------------------------------------------------
extern "C" void kernel_launch(void* const* d_in, const int* in_sizes, int n_in,
                              void* d_out, int out_size, void* d_ws, size_t ws_size,
                              hipStream_t stream) {
    const float* query  = (const float*)d_in[0];   // (16,1024)
    const float* states = (const float*)d_in[1];   // (16,2048,1024)
    const float* Wq     = (const float*)d_in[2];   // (1024,1024)
    const float* bq     = (const float*)d_in[3];   // (1024,)
    const float* Wk     = (const float*)d_in[4];   // (1024,1024)
    const float* bk     = (const float*)d_in[5];   // (1024,)
    const float* We     = (const float*)d_in[6];   // (2048,)
    const float* be     = (const float*)d_in[7];   // (1,)
    const float* we_q = We;
    const float* we_k = We + 1024;

    float* out_w    = (float*)d_out;               // attn_weights: 16*2048
    float* out_attn = (float*)d_out + 16 * 2048;   // attn: 16*1024

    // workspace layout (bytes)
    char* ws = (char*)d_ws;
    ushort_t* wk_swz = (ushort_t*)ws;                         // 2 MB
    float* eq   = (float*)(ws + (2u << 20));                  // 64 B
    float* ek2  = (float*)(ws + (2u << 20) + 1024);           // 256 KB
    float* part = (float*)(ws + (2u << 20) + 1024 + 262144);  // 1 MB

    wk_convert_kernel<<<4096, 256, 0, stream>>>(Wk, wk_swz);
    qpath_kernel<<<16, 256, 0, stream>>>(query, Wq, bq, we_q, eq);
    energy_kernel<<<512, 256, 256 * 1024, stream>>>(states, wk_swz, bk, we_k, ek2);
    softmax_kernel<<<16, 256, 0, stream>>>(ek2, eq, be, out_w);
    attn_partial_kernel<<<256, 256, 0, stream>>>(out_w, states, part);
    attn_reduce_kernel<<<64, 256, 0, stream>>>(part, out_attn);
}
